// HypFormer_64536178589696
// MI455X (gfx1250) — matrix operand order repaired and checked by reading, not compiled, not run on hardware
//
#include <hip/hip_runtime.h>

// ---------------------------------------------------------------------------
// HypFormer forward for MI455X (gfx1250, wave32, WMMA bf16 path)
// Double-buffered LDS pipeline: stage tile s+1 while WMMA consumes tile s.
// ---------------------------------------------------------------------------

typedef __attribute__((ext_vector_type(16))) __bf16 v16bf;
typedef __attribute__((ext_vector_type(8)))  float  v8f;

#define TM 64
#define TN 128
#define TK 32
#define ASTR 40   // LDS row stride (ushorts): 80B rows -> 16B-aligned sublanes
#define BSTR 40
#define F_RELU  1
#define F_ACCUM 2

__device__ __forceinline__ unsigned short f2bf(float f) {
  union { float f; unsigned u; } v; v.f = f;
  unsigned r = v.u + 0x7FFFu + ((v.u >> 16) & 1u);   // round-to-nearest-even
  return (unsigned short)(r >> 16);
}

union FragU { struct { uint4 a, b; } q; v16bf v; };

// --- branch-free staging: 8 elems base[(k0+j)*stride] -> bf16 -> ds_store_b128
__device__ __forceinline__ void stage8_full(const float* __restrict__ base,
                                            int stride, int k0,
                                            unsigned short* dst) {
  float t[8];
#pragma unroll
  for (int j = 0; j < 8; ++j) t[j] = base[(size_t)(k0 + j) * stride];
  uint4 p; unsigned short* ps = (unsigned short*)&p;
#pragma unroll
  for (int j = 0; j < 8; ++j) ps[j] = f2bf(t[j]);
  *(uint4*)dst = p;
}

// tail variant: clamp address (stays in-bounds), zero k>=klim via {0,1} multiply
__device__ __forceinline__ void stage8_tail(const float* __restrict__ base,
                                            int stride, int k0, int klim,
                                            unsigned short* dst) {
  float t[8];
#pragma unroll
  for (int j = 0; j < 8; ++j) {
    const int gk  = k0 + j;
    const int gkc = gk < klim ? gk : klim - 1;
    t[j] = base[(size_t)gkc * stride];
  }
#pragma unroll
  for (int j = 0; j < 8; ++j) t[j] *= (k0 + j < klim) ? 1.f : 0.f;
  uint4 p; unsigned short* ps = (unsigned short*)&p;
#pragma unroll
  for (int j = 0; j < 8; ++j) ps[j] = f2bf(t[j]);
  *(uint4*)dst = p;
}

// Load A frag + all 4 B frags into distinct registers, then 4 WMMAs.
__device__ __forceinline__ void wmma_step(const unsigned short* As,
                                          const unsigned short* Bs,
                                          int mw, int nw, int hf, int r,
                                          v8f acc[4]) {
  FragU af, bf0, bf1, bf2, bf3;
  const int ab = (mw + r) * ASTR + hf * 16;    // A: contiguous K half
  af.q.a = *(const uint4*)&As[ab];
  af.q.b = *(const uint4*)&As[ab + 8];
  const int bb = (nw + r) * BSTR + hf * 16;    // B: contiguous K half
  bf0.q.a = *(const uint4*)&Bs[bb];
  bf0.q.b = *(const uint4*)&Bs[bb + 8];
  bf1.q.a = *(const uint4*)&Bs[bb + 16 * BSTR];
  bf1.q.b = *(const uint4*)&Bs[bb + 16 * BSTR + 8];
  bf2.q.a = *(const uint4*)&Bs[bb + 32 * BSTR];
  bf2.q.b = *(const uint4*)&Bs[bb + 32 * BSTR + 8];
  bf3.q.a = *(const uint4*)&Bs[bb + 48 * BSTR];
  bf3.q.b = *(const uint4*)&Bs[bb + 48 * BSTR + 8];
  acc[0] = __builtin_amdgcn_wmma_f32_16x16x32_bf16(false, af.v, false, bf0.v,
                                                   (short)0, acc[0], false, false);
  acc[1] = __builtin_amdgcn_wmma_f32_16x16x32_bf16(false, af.v, false, bf1.v,
                                                   (short)0, acc[1], false, false);
  acc[2] = __builtin_amdgcn_wmma_f32_16x16x32_bf16(false, af.v, false, bf2.v,
                                                   (short)0, acc[2], false, false);
  acc[3] = __builtin_amdgcn_wmma_f32_16x16x32_bf16(false, af.v, false, bf3.v,
                                                   (short)0, acc[3], false, false);
}

// ---------------------------------------------------------------------------
// Canonical GEMM: C[M,Nc] = epilogue( A[M,K] * B ),  A fp32 row-major.
// bmode==0: B[k][n] = Bsrc[n*K + k]   (weight W[out][in] -> x @ W^T)
// bmode==1: B[k][n] = Bsrc[k*Nc + n]  (K-major source, e.g. ktv^T)
// A rows >= M / B cols >= Nc are staged from clamped (valid) addresses: they
// only influence discarded C rows/cols. Only the K tail (K=257) is masked.
// ---------------------------------------------------------------------------
__global__ __launch_bounds__(256) void gemm_wmma_kernel(
    const float* __restrict__ A, int lda, int K, int M,
    const float* __restrict__ Bsrc, int bmode, int Nc,
    const float* __restrict__ bias, const float* __restrict__ resid, int ldr,
    float* __restrict__ C, int ldc, int coff, float alpha, int flags)
{
  __shared__ unsigned short As[2][TM * ASTR];
  __shared__ unsigned short Bs[2][TN * BSTR];
  const int tid  = threadIdx.x;
  const int lane = tid & 31;
  const int wid  = tid >> 5;
  const int hf   = lane >> 4;        // wave half (K-block select)
  const int r    = lane & 15;
  const int m0   = blockIdx.y * TM;
  const int n0   = blockIdx.x * TN;
  const int mw   = (wid & 3) * 16;   // 4 row-groups of 16
  const int nw   = (wid >> 2) * 64;  // 2 col-groups of 64 (4 accs each)
  const int sma = tid >> 2, ska = (tid & 3) * 8;   // A: 8 elems/thread
  const int snb = tid >> 1, skb = (tid & 1) * 16;  // B: 16 elems/thread

  v8f acc[4];
#pragma unroll
  for (int a = 0; a < 4; ++a) acc[a] = (v8f){0.f,0.f,0.f,0.f,0.f,0.f,0.f,0.f};

  // clamped base addresses (always valid); no row/col masks needed
  const int gm  = m0 + sma;
  const int gmc = gm < M ? gm : M - 1;
  const float* Abase = A + (size_t)gmc * lda;      // stride 1 in k
  const int gn  = n0 + snb;
  const int gnc = gn < Nc ? gn : Nc - 1;
  const float* Bbase = (bmode == 0) ? Bsrc + (size_t)gnc * K : Bsrc + gnc;
  const int bstride  = (bmode == 0) ? 1 : Nc;

  const int adst = sma * ASTR + ska;
  const int bdst = snb * BSTR + skb;

  auto stage = [&](int s, int b) {
    const int kb = s * TK;
    if (kb + TK <= K) {
      stage8_full(Abase, 1, kb + ska, &As[b][adst]);
      stage8_full(Bbase, bstride, kb + skb,     &Bs[b][bdst]);
      stage8_full(Bbase, bstride, kb + skb + 8, &Bs[b][bdst + 8]);
    } else {
      stage8_tail(Abase, 1, kb + ska, K, &As[b][adst]);
      stage8_tail(Bbase, bstride, kb + skb,     K, &Bs[b][bdst]);
      stage8_tail(Bbase, bstride, kb + skb + 8, K, &Bs[b][bdst + 8]);
    }
  };

  const int nsteps = (K + TK - 1) / TK;
  stage(0, 0);
  __syncthreads();
  int cur = 0;
  for (int s = 0; s < nsteps; ++s) {
    if (s + 1 < nsteps) stage(s + 1, cur ^ 1);   // overlap next-tile loads
    if (s + 2 < nsteps) {                        // cache hint two tiles ahead
      const int pk = (s + 2) * TK + ska;
      __builtin_prefetch(&Abase[pk < K ? pk : K - 1], 0, 1);
    }
    wmma_step(As[cur], Bs[cur], mw, nw, hf, r, acc);
    __syncthreads();                             // one barrier per K-step
    cur ^= 1;
  }

  // epilogue: C/D layout -> row = base + v + 8*hf, col = base + r
  const int rowb = m0 + mw + hf * 8;
#pragma unroll
  for (int a = 0; a < 4; ++a) {
    const int col = n0 + nw + a * 16 + r;
    if (col >= Nc) continue;
    const float bv = bias ? bias[col] : 0.f;
#pragma unroll
    for (int v = 0; v < 8; ++v) {
      const int gr = rowb + v;
      if (gr >= M) continue;
      float t = acc[a][v] + bv;
      if (flags & F_RELU) t = fmaxf(t, 0.f);
      t *= alpha;
      if (resid) t += resid[(size_t)gr * ldr + col];
      float* cp = &C[(size_t)gr * ldc + coff + col];
      if (flags & F_ACCUM) t += *cp;
      *cp = t;
    }
  }
}

// ---------------------------------------------------------------------------
// Split-K reduction GEMM: ktvT[d][h] += sum_n phi_k[n][d] * v[n][h]
// (transposed gathers, stride 256; K = N = 50000, M = Nc = 256)
// grid.x = 8 tiles (4 m-tiles x 2 n-tiles), grid.y = K chunks.
// ---------------------------------------------------------------------------
__global__ __launch_bounds__(256) void gemm_ktv_kernel(
    const float* __restrict__ Pk, const float* __restrict__ V,
    float* __restrict__ Cout, int Ntot, int kchunk)
{
  __shared__ unsigned short As[2][TM * ASTR];
  __shared__ unsigned short Bs[2][TN * BSTR];
  const int tid = threadIdx.x, lane = tid & 31, wid = tid >> 5;
  const int hf = lane >> 4, r = lane & 15;
  const int m0 = (blockIdx.x & 3) * TM;    // d-tile
  const int n0 = (blockIdx.x >> 2) * TN;   // h-tile
  const int mw = (wid & 3) * 16, nw = (wid >> 2) * 64;
  const int sma = tid >> 2, ska = (tid & 3) * 8;
  const int snb = tid >> 1, skb = (tid & 1) * 16;
  const int kstart = blockIdx.y * kchunk;
  int kend = kstart + kchunk; if (kend > Ntot) kend = Ntot;

  v8f acc[4];
#pragma unroll
  for (int a = 0; a < 4; ++a) acc[a] = (v8f){0.f,0.f,0.f,0.f,0.f,0.f,0.f,0.f};

  const float* Abase = Pk + (m0 + sma);    // stride 256 in k (transposed)
  const float* Bbase = V  + (n0 + snb);    // stride 256 in k
  const int adst = sma * ASTR + ska;
  const int bdst = snb * BSTR + skb;

  auto stage = [&](int s, int b) {
    const int kb = kstart + s * TK;
    if (kb + TK <= kend) {
      stage8_full(Abase, 256, kb + ska,     &As[b][adst]);
      stage8_full(Bbase, 256, kb + skb,     &Bs[b][bdst]);
      stage8_full(Bbase, 256, kb + skb + 8, &Bs[b][bdst + 8]);
    } else {
      stage8_tail(Abase, 256, kb + ska,     kend, &As[b][adst]);
      stage8_tail(Bbase, 256, kb + skb,     kend, &Bs[b][bdst]);
      stage8_tail(Bbase, 256, kb + skb + 8, kend, &Bs[b][bdst + 8]);
    }
  };

  const int span = kend - kstart;
  const int nsteps = (span + TK - 1) / TK;
  if (nsteps > 0) stage(0, 0);
  __syncthreads();
  int cur = 0;
  for (int s = 0; s < nsteps; ++s) {
    if (s + 1 < nsteps) stage(s + 1, cur ^ 1);
    wmma_step(As[cur], Bs[cur], mw, nw, hf, r, acc);
    __syncthreads();
    cur ^= 1;
  }

  const int rowb = m0 + mw + hf * 8;
#pragma unroll
  for (int a = 0; a < 4; ++a) {
    const int col = n0 + nw + a * 16 + r;
#pragma unroll
    for (int v = 0; v < 8; ++v)
      atomicAdd(&Cout[(size_t)(rowb + v) * 256 + col], acc[a][v]);
  }
}

// ---------------------------------------------------------------------------
// Elementwise / reduction kernels (block = 256 threads = one node)
// ---------------------------------------------------------------------------
__device__ __forceinline__ float blockReduceSum(float v, float* sh) {
  const int t = threadIdx.x;
  sh[t] = v; __syncthreads();
  for (int s = 128; s > 0; s >>= 1) { if (t < s) sh[t] += sh[t + s]; __syncthreads(); }
  float r = sh[0]; __syncthreads();
  return r;
}

// xe = expmap0(concat(1, x), k=1): xe[0]=cosh(|x|), xe[1+j]=sinh(|x|)*x_j/|x|
__global__ __launch_bounds__(256) void expmap_kernel(const float* __restrict__ x,
                                                     float* __restrict__ xe) {
  __shared__ float sh[256];
  const int n = blockIdx.x, t = threadIdx.x;
  const float v = x[(size_t)n * 256 + t];
  const float ss = blockReduceSum(v * v, sh);
  const float nr = sqrtf(fmaxf(ss, 1e-8f));
  if (t == 0) xe[(size_t)n * 257] = coshf(nr);
  xe[(size_t)n * 257 + 1 + t] = sinhf(nr) * v / nr;
}

// x[n][0] = sqrt(sum_j x[n][1+j]^2 + k)
__global__ __launch_bounds__(256) void settime_kernel(float* __restrict__ xh, float k) {
  __shared__ float sh[256];
  const int n = blockIdx.x, t = threadIdx.x;
  const float v = xh[(size_t)n * 257 + 1 + t];
  const float ss = blockReduceSum(v * v, sh);
  if (t == 0) xh[(size_t)n * 257] = sqrtf(ss + k);
}

// out = add_time(relu(LN(midpoint(0.5*(a+b)) space part)*g+b), k)
__global__ __launch_bounds__(256) void mid_ln_relu_kernel(
    const float* __restrict__ a, const float* __restrict__ b,
    const float* __restrict__ gamma, const float* __restrict__ beta,
    float* __restrict__ out, float k) {
  __shared__ float sh[256];
  const int n = blockIdx.x, t = threadIdx.x;
  const size_t ra = (size_t)n * 257;
  const float avt = 0.5f * (a[ra] + b[ra]);
  const float av  = 0.5f * (a[ra + 1 + t] + b[ra + 1 + t]);
  const float ss  = blockReduceSum(av * av, sh);
  const float li  = ss - avt * avt;                 // l_inner(ave, ave)
  const float den = sqrtf(fmaxf(fabsf(li), 1e-8f));
  const float m   = sqrtf(k) * av / den;            // midpoint (space part)
  const float mu  = blockReduceSum(m, sh) * (1.f / 256.f);
  const float d   = m - mu;
  const float var = blockReduceSum(d * d, sh) * (1.f / 256.f);
  const float xs  = d * rsqrtf(var + 1e-5f) * gamma[t] + beta[t];
  const float rl  = fmaxf(xs, 0.f);
  const float ts  = blockReduceSum(rl * rl, sh);
  if (t == 0) out[ra] = sqrtf(ts + k);
  out[ra + 1 + t] = rl;
}

// in-place phi: y=(relu(q)+eps)/|ns|; q = sqrt(S1)/sqrt(S2) * y^2
__global__ __launch_bounds__(256) void phi_kernel(float* __restrict__ q,
                                                  const float* __restrict__ nsp) {
  __shared__ float sh[256];
  const int n = blockIdx.x, t = threadIdx.x;
  const float ns = fabsf(nsp[0]);
  const float y  = (fmaxf(q[(size_t)n * 256 + t], 0.f) + 1e-6f) / ns;
  const float y2 = y * y;
  const float s1 = blockReduceSum(y2, sh);
  const float s2 = blockReduceSum(y2 * y2, sh);
  q[(size_t)n * 256 + t] = sqrtf(s1) / sqrtf(s2) * y2;
}

__global__ __launch_bounds__(256) void colsum_kernel(const float* __restrict__ pk,
                                                     float* __restrict__ s, int N) {
  const int t = threadIdx.x;
  float acc = 0.f;
  for (int n = blockIdx.x; n < N; n += gridDim.x) acc += pk[(size_t)n * 256 + t];
  atomicAdd(&s[t], acc);
}

__global__ __launch_bounds__(256) void den_kernel(const float* __restrict__ q,
                                                  const float* __restrict__ s,
                                                  float* __restrict__ den) {
  __shared__ float sh[256];
  const int n = blockIdx.x, t = threadIdx.x;
  const float v = q[(size_t)n * 256 + t] * s[t];
  const float ss = blockReduceSum(v, sh);
  if (t == 0) den[n] = ss;
}

// attn = num/(den+eps)+vmap; out = midpoint(add_time(attn)) (full 257 vec)
__global__ __launch_bounds__(256) void trans_post_kernel(
    const float* __restrict__ num, const float* __restrict__ den,
    const float* __restrict__ vmap, float* __restrict__ out, float k) {
  __shared__ float sh[256];
  const int n = blockIdx.x, t = threadIdx.x;
  const float attn = num[(size_t)n * 256 + t] / (den[n] + 1e-6f)
                   + vmap[(size_t)n * 256 + t];
  const float ss = blockReduceSum(attn * attn, sh);
  const float tt = sqrtf(ss + k);
  const float li = ss - tt * tt;
  const float dn = sqrtf(fmaxf(fabsf(li), 1e-8f));
  const float sk = sqrtf(k);
  if (t == 0) out[(size_t)n * 257] = sk * tt / dn;
  out[(size_t)n * 257 + 1 + t] = sk * attn / dn;
}

__global__ __launch_bounds__(256) void zero_kernel(float* __restrict__ p, size_t n) {
  const size_t i = (size_t)blockIdx.x * 256 + threadIdx.x;
  if (i < n) p[i] = 0.f;
}

__global__ __launch_bounds__(256) void deg_kernel(const int* __restrict__ col,
                                                  float* __restrict__ deg, int E) {
  const int i = blockIdx.x * 256 + threadIdx.x;
  if (i < E) atomicAdd(&deg[col[i]], 1.f);
}

__global__ __launch_bounds__(256) void norm_kernel(const float* __restrict__ deg,
                                                   float* __restrict__ nrm, int N) {
  const int i = blockIdx.x * 256 + threadIdx.x;
  if (i < N) {
    const float d = deg[i];
    nrm[i] = (d > 0.f) ? rsqrtf(fmaxf(d, 1.f)) : 0.f;
  }
}

__global__ __launch_bounds__(256) void scatter_kernel(
    const float* __restrict__ h, const int* __restrict__ row,
    const int* __restrict__ col, const float* __restrict__ nrm,
    float* __restrict__ agg, int E) {
  const int t = threadIdx.x;
  for (int e = blockIdx.x; e < E; e += gridDim.x) {
    const int rs = row[e], cs = col[e];
    const float coef = nrm[rs] * nrm[cs];
    atomicAdd(&agg[(size_t)cs * 256 + t], h[(size_t)rs * 256 + t] * coef);
  }
}

// ---------------------------------------------------------------------------
// Host orchestration
// ---------------------------------------------------------------------------
extern "C" void kernel_launch(void* const* d_in, const int* in_sizes, int n_in,
                              void* d_out, int out_size, void* d_ws, size_t ws_size,
                              hipStream_t stream) {
  (void)n_in; (void)out_size; (void)ws_size;
  const int N = in_sizes[0] / 256;
  const int E = in_sizes[1] / 2;
  const float* x  = (const float*)d_in[0];
  const int* ei   = (const int*)d_in[1];
  const int* erow = ei;
  const int* ecol = ei + E;

  // params flattened in dict order
  int pi = 2;
  auto P = [&](int i) { return (const float*)d_in[i]; };
  const float *W_fcin = P(pi), *b_fcin = P(pi + 1); pi += 2;
  const float *W_pos  = P(pi), *b_pos  = P(pi + 1); pi += 2;
  const float *lnG[3], *lnB[3];
  for (int i = 0; i < 3; ++i) { lnG[i] = P(pi); lnB[i] = P(pi + 1); pi += 2; }
  struct Lay { const float *Wq,*bq,*Wk,*bk,*Wv,*bv,*Wm,*bm,*ns; } L[2];
  for (int i = 0; i < 2; ++i) {
    L[i].Wq = P(pi); L[i].bq = P(pi+1); L[i].Wk = P(pi+2); L[i].bk = P(pi+3);
    L[i].Wv = P(pi+4); L[i].bv = P(pi+5); L[i].Wm = P(pi+6); L[i].bm = P(pi+7);
    L[i].ns = P(pi+8); pi += 9;
  }
  const float *W_fco = P(pi), *b_fco = P(pi + 1); pi += 2;
  const float *W_g0  = P(pi), *b_g0  = P(pi + 1); pi += 2;
  const float *W_g1  = P(pi), *b_g1  = P(pi + 1); pi += 2;
  const float *W_g2  = P(pi), *b_g2  = P(pi + 1); pi += 2;
  const float *W_dt  = P(pi), *b_dt  = P(pi + 1); pi += 2;
  const float *W_dg  = P(pi), *b_dg  = P(pi + 1); pi += 2;

  // workspace carve-up
  float* w = (float*)d_ws;
  size_t off = 0;
  auto alloc = [&](size_t n) { float* p = w + off; off += n; return p; };
  float* XE  = alloc((size_t)N * 257);
  float* X0  = alloc((size_t)N * 257);
  float* X1  = alloc((size_t)N * 257);
  float* X2  = alloc((size_t)N * 257);
  float* Q   = alloc((size_t)N * 256);
  float* Kb  = alloc((size_t)N * 256);
  float* Vb  = alloc((size_t)N * 256);
  float* NU  = alloc((size_t)N * 256);
  float* VM  = alloc((size_t)N * 256);
  float* DEN = alloc(N);
  float* DEG = alloc(N);
  float* NRM = alloc(N);
  float* SV  = alloc(256);
  float* KTV = alloc(256 * 256);

  auto gemm = [&](const float* A, int lda, int K, int M,
                  const float* B, int bmode, int Nc,
                  const float* bias, const float* resid, int ldr,
                  float* C, int ldc, int coff, float alpha, int flags) {
    dim3 grid((Nc + TN - 1) / TN, (M + TM - 1) / TM);
    hipLaunchKernelGGL(gemm_wmma_kernel, grid, dim3(256), 0, stream,
                       A, lda, K, M, B, bmode, Nc, bias, resid, ldr,
                       C, ldc, coff, alpha, flags);
  };
  auto zero = [&](float* p, size_t n) {
    hipLaunchKernelGGL(zero_kernel, dim3((unsigned)((n + 255) / 256)), dim3(256),
                       0, stream, p, n);
  };

  // ---- trans_conv ----
  hipLaunchKernelGGL(expmap_kernel, dim3(N), dim3(256), 0, stream, x, XE);
  gemm(XE, 257, 257, N, W_fcin, 0, 256, b_fcin, nullptr, 256, X0, 257, 1, 1.f, 0);
  hipLaunchKernelGGL(settime_kernel, dim3(N), dim3(256), 0, stream, X0, 1.f);
  gemm(XE, 257, 257, N, W_pos, 0, 256, b_pos, nullptr, 256, X1, 257, 1, 1.f, 0);
  hipLaunchKernelGGL(settime_kernel, dim3(N), dim3(256), 0, stream, X1, 1.f);
  hipLaunchKernelGGL(mid_ln_relu_kernel, dim3(N), dim3(256), 0, stream,
                     X0, X1, lnG[0], lnB[0], X2, 1.f);

  float* xcur = X2; float* tbuf = X0; float* mbuf = X1;
  for (int i = 0; i < 2; ++i) {
    gemm(xcur, 257, 257, N, L[i].Wq, 0, 256, L[i].bq, nullptr, 256, Q,  256, 0, 1.f, 0);
    gemm(xcur, 257, 257, N, L[i].Wk, 0, 256, L[i].bk, nullptr, 256, Kb, 256, 0, 1.f, 0);
    gemm(xcur, 257, 257, N, L[i].Wv, 0, 256, L[i].bv, nullptr, 256, Vb, 256, 0, 1.f, 0);
    hipLaunchKernelGGL(phi_kernel, dim3(N), dim3(256), 0, stream, Q,  L[i].ns);
    hipLaunchKernelGGL(phi_kernel, dim3(N), dim3(256), 0, stream, Kb, L[i].ns);
    zero(SV, 256);
    hipLaunchKernelGGL(colsum_kernel, dim3(128), dim3(256), 0, stream, Kb, SV, N);
    zero(KTV, 256 * 256);
    const int kch = 4096;
    dim3 gk(8, (unsigned)((N + kch - 1) / kch));
    hipLaunchKernelGGL(gemm_ktv_kernel, gk, dim3(256), 0, stream, Kb, Vb, KTV, N, kch);
    gemm(Q, 256, 256, N, KTV, 1, 256, nullptr, nullptr, 256, NU, 256, 0, 1.f, 0);
    hipLaunchKernelGGL(den_kernel, dim3(N), dim3(256), 0, stream, Q, SV, DEN);
    gemm(Vb, 256, 256, N, L[i].Wm, 0, 256, L[i].bm, nullptr, 256, VM, 256, 0, 1.f, 0);
    hipLaunchKernelGGL(trans_post_kernel, dim3(N), dim3(256), 0, stream,
                       NU, DEN, VM, tbuf, 1.f);
    hipLaunchKernelGGL(mid_ln_relu_kernel, dim3(N), dim3(256), 0, stream,
                       tbuf, xcur, lnG[i + 1], lnB[i + 1], mbuf, 1.f);
    float* tmp = xcur; xcur = mbuf; mbuf = tmp;   // tbuf reused next iter
  }
  float* XT = tbuf;  // x1 = fc_out(xcur)
  gemm(xcur, 257, 257, N, W_fco, 0, 256, b_fco, nullptr, 256, XT, 257, 1, 1.f, 0);
  hipLaunchKernelGGL(settime_kernel, dim3(N), dim3(256), 0, stream, XT, 1.f);

  // ---- graph_conv (reuse Q/Kb/Vb as H1/H2/AGG) ----
  float* H1 = Q; float* H2 = Kb; float* AG = Vb;
  gemm(x, 256, 256, N, W_g0, 0, 256, b_g0, nullptr, 256, H1, 256, 0, 1.f, F_RELU);
  zero(DEG, (size_t)N);
  hipLaunchKernelGGL(deg_kernel, dim3((E + 255) / 256), dim3(256), 0, stream, ecol, DEG, E);
  hipLaunchKernelGGL(norm_kernel, dim3((N + 255) / 256), dim3(256), 0, stream, DEG, NRM, N);
  zero(AG, (size_t)N * 256);
  hipLaunchKernelGGL(scatter_kernel, dim3(E), dim3(256), 0, stream, H1, erow, ecol, NRM, AG, E);
  gemm(AG, 256, 256, N, W_g1, 0, 256, b_g1, H1, 256, H2, 256, 0, 1.f, F_RELU);
  zero(AG, (size_t)N * 256);
  hipLaunchKernelGGL(scatter_kernel, dim3(E), dim3(256), 0, stream, H2, erow, ecol, NRM, AG, E);
  gemm(AG, 256, 256, N, W_g2, 0, 256, b_g2, H2, 256, H1, 256, 0, 1.f, F_RELU);

  // ---- decoders: out = 0.5*(x1@Wt^T+bt) + 0.5*(h@Wg^T+bg) ----
  float* outp = (float*)d_out;
  gemm(XT, 257, 257, N, W_dt, 0, 128, b_dt, nullptr, 256, outp, 128, 0, 0.5f, 0);
  gemm(H1, 256, 256, N, W_dg, 0, 128, b_dg, nullptr, 256, outp, 128, 0, 0.5f, F_ACCUM);
}
